// EncoDecLSTM_49271864820090
// MI455X (gfx1250) — matrix operand order repaired and checked by reading, not compiled, not run on hardware
//
#include <hip/hip_runtime.h>

// ---------------------------------------------------------------------------
// CDNA5 WMMA (wave32): D(f32 16x16) = A(bf16 16x32) * B(bf16 32x16) + C
// ---------------------------------------------------------------------------
typedef __attribute__((ext_vector_type(16))) __bf16 v16bf;
typedef __attribute__((ext_vector_type(8)))  __bf16 v8bf;
typedef __attribute__((ext_vector_type(8)))  float  v8f;

#define B_  256
#define T_  512
#define F_  64
#define U_  128

// Packed-weight fragment regions (fragment = 32 lanes x 16 bf16 = 1 KB)
#define FRAG_ENC_K   0     // 32 ntiles * 2 ktiles  = 64
#define FRAG_ENC_R   64    // 32 ntiles * 4 ktiles  = 128
#define FRAG_DEC_SUM 192   // 32 ntiles * 4 ktiles  = 128  (dec_k + dec_rk)
#define FRAG_W1      320   // 8 ntiles * 4 ktiles   = 32
#define FRAG_W2      352   // 4 ntiles * 4 ktiles   = 16
#define FRAG_TOTAL   368

#define HP 136   // LDS pitch (halfs) for 128-wide tiles
#define XP 72    // LDS pitch (halfs) for 64-wide x tile

__device__ __forceinline__ float sigmoidf_(float v) {
    return 1.f / (1.f + __expf(-v));
}
__device__ __forceinline__ float tanhf_(float v) {
    return 1.f - 2.f / (__expf(2.f * v) + 1.f);
}

// A-matrix bf16 16x32 fragment from a row-major bf16 tile (pitch in halfs).
// Lane L (m=L%16): elems 0..15 -> k=base..base+15,
// base = kt*32 + (L>=16 ? 16 : 0)  => two 16-byte loads per lane.
__device__ __forceinline__ v16bf load_a_frag(const __bf16* tile, int pitch, int kt) {
    const int lane = threadIdx.x & 31;
    const int m    = lane & 15;
    const int base = kt * 32 + ((lane & 16) ? 16 : 0);
    const __bf16* p = tile + m * pitch + base;
    union { v16bf v; v8bf h[2]; } u;
    u.h[0] = *(const v8bf*)(p);
    u.h[1] = *(const v8bf*)(p + 8);
    return u.v;
}

// B-matrix fragment from packed global buffer: frag-major, lane-major, 16 halfs.
__device__ __forceinline__ v16bf load_w_frag(const __bf16* wpk, int fragId) {
    const int lane = threadIdx.x & 31;
    const __bf16* p = wpk + (size_t)fragId * 512 + lane * 16;
    union { v16bf v; v8bf h[2]; } u;
    u.h[0] = *(const v8bf*)(p);
    u.h[1] = *(const v8bf*)(p + 8);
    return u.v;
}

__device__ __forceinline__ v8f splat8(float s) {
    v8f r;
#pragma unroll
    for (int i = 0; i < 8; ++i) r[i] = s;
    return r;
}

__device__ __forceinline__ v8f wmma_bf16(v16bf a, v16bf b, v8f c) {
    return __builtin_amdgcn_wmma_f32_16x16x32_bf16(false, a, false, b, (short)0, c,
                                                   false, false);
}

// ---------------------------------------------------------------------------
// Kernel 1: f32 weights -> bf16, decoder pair pre-summed, packed into WMMA
// B-fragment layout: lane L holds column n = nt*16 + L%16,
// elems e: k = kt*32 + (L>=16 ? 16 : 0) + e.  One thread per (fragment, lane).
// ---------------------------------------------------------------------------
__global__ void pack_weights_kernel(const float* __restrict__ enc_k,
                                    const float* __restrict__ enc_rk,
                                    const float* __restrict__ dec_k,
                                    const float* __restrict__ dec_rk,
                                    const float* __restrict__ w1,
                                    const float* __restrict__ w2,
                                    __bf16* __restrict__ out) {
    int gid = blockIdx.x * blockDim.x + threadIdx.x;
    if (gid >= FRAG_TOTAL * 32) return;
    int frag = gid >> 5;
    int lane = gid & 31;

    const float* S0 = nullptr;
    const float* S1 = nullptr;
    int N = 0, kTiles = 0, fbase = 0;
    if (frag < FRAG_ENC_R)        { S0 = enc_k;  N = 512; kTiles = 2; fbase = FRAG_ENC_K; }
    else if (frag < FRAG_DEC_SUM) { S0 = enc_rk; N = 512; kTiles = 4; fbase = FRAG_ENC_R; }
    else if (frag < FRAG_W1)      { S0 = dec_k; S1 = dec_rk; N = 512; kTiles = 4; fbase = FRAG_DEC_SUM; }
    else if (frag < FRAG_W2)      { S0 = w1;     N = 128; kTiles = 4; fbase = FRAG_W1; }
    else                          { S0 = w2;     N = 64;  kTiles = 4; fbase = FRAG_W2; }

    int lf = frag - fbase;
    int nt = lf / kTiles;
    int kt = lf % kTiles;
    int n  = nt * 16 + (lane & 15);
    int kb = kt * 32 + ((lane & 16) ? 16 : 0);

    v8bf lo, hi;
#pragma unroll
    for (int e = 0; e < 8; ++e) {
        float a = S0[(size_t)(kb + e) * N + n];
        if (S1) a += S1[(size_t)(kb + e) * N + n];
        lo[e] = (__bf16)a;
    }
#pragma unroll
    for (int e = 0; e < 8; ++e) {
        float a = S0[(size_t)(kb + 8 + e) * N + n];
        if (S1) a += S1[(size_t)(kb + 8 + e) * N + n];
        hi[e] = (__bf16)a;
    }
    __bf16* dst = out + (size_t)frag * 512 + lane * 16;
    *(v8bf*)(dst)     = lo;
    *(v8bf*)(dst + 8) = hi;
}

// ---------------------------------------------------------------------------
// Kernel 2 (optional, full-chip GEMM): zx[t] = x_t @ enc_k, stored as bf16 in
// C-layout fragments: frag = ((t*16+bt)*8+wv)*4+g, lane-major 8 halfs (16 B).
// grid = (T, 16), 256 threads.
// ---------------------------------------------------------------------------
__global__ __launch_bounds__(256)
void zx_precompute_kernel(const float* __restrict__ x,
                          const __bf16* __restrict__ wpk,
                          __bf16* __restrict__ zxp) {
    __shared__ __align__(16) __bf16 xs[16 * XP];
    const int tid  = threadIdx.x;
    const int lane = tid & 31;
    const int wv   = tid >> 5;
    const int t    = blockIdx.x;
    const int bt   = blockIdx.y;
    const int b0   = bt * 16;

    {   // stage x_t [16 x 64] tile
        const int i  = tid >> 4;
        const int k0 = (tid & 15) * 4;
        const float4 v = *(const float4*)(x + ((size_t)(b0 + i) * T_ + t) * F_ + k0);
        __bf16* d = xs + i * XP + k0;
        d[0] = (__bf16)v.x; d[1] = (__bf16)v.y;
        d[2] = (__bf16)v.z; d[3] = (__bf16)v.w;
    }
    __syncthreads();

    v16bf wk[8];
#pragma unroll
    for (int g = 0; g < 4; ++g)
#pragma unroll
        for (int kt = 0; kt < 2; ++kt)
            wk[g * 2 + kt] = load_w_frag(wpk, FRAG_ENC_K + (g * 8 + wv) * 2 + kt);

    v8f acc[4];
#pragma unroll
    for (int g = 0; g < 4; ++g) acc[g] = splat8(0.f);
#pragma unroll
    for (int kt = 0; kt < 2; ++kt) {
        v16bf a = load_a_frag(xs, XP, kt);
#pragma unroll
        for (int g = 0; g < 4; ++g) acc[g] = wmma_bf16(a, wk[g * 2 + kt], acc[g]);
    }
#pragma unroll
    for (int g = 0; g < 4; ++g) {
        v8bf o;
#pragma unroll
        for (int r = 0; r < 8; ++r) o[r] = (__bf16)acc[g][r];
        size_t fi = (((size_t)t * 16 + bt) * 8 + wv) * 4 + g;
        *(v8bf*)(zxp + fi * 256 + lane * 8) = o;
    }
}

// ---------------------------------------------------------------------------
// Kernel 3: persistent encoder/decoder LSTM.
// grid = B/16 workgroups; 8 wave32s; wave w owns hidden cols [w*16, w*16+16).
// USE_ZX:     encoder reads precomputed x@enc_k fragments (prefetched 1 step
//             ahead) instead of staging x and doing K=64 WMMAs.
// FUSE_DENSE: dense head fused per decoder step (min-scratch fallback);
//             otherwise h is scattered to the seq buffer for a later kernel.
// ---------------------------------------------------------------------------
template <bool USE_ZX, bool FUSE_DENSE>
__global__ __launch_bounds__(256)
void lstm_encdec_kernel(const float* __restrict__ x,
                        const float* __restrict__ enc_b,
                        const float* __restrict__ dec_b,
                        const float* __restrict__ b1,
                        const float* __restrict__ b2,
                        const __bf16* __restrict__ wpk,
                        const __bf16* __restrict__ zxp,
                        __bf16* __restrict__ seq,
                        float* __restrict__ out) {
    __shared__ __align__(16) __bf16 hs[16 * HP];        // h tile [16 x 128]
    __shared__ __align__(16) __bf16 xs[2][16 * XP];     // double-buffered x tile
    __shared__ __align__(16) __bf16 dsb[16 * HP];       // hidden tile (fused dense)

    const int tid     = threadIdx.x;
    const int lane    = tid & 31;
    const int wv      = tid >> 5;
    const int bt      = blockIdx.x;
    const int b0      = bt * 16;
    const int col     = lane & 15;
    const int rowBase = (lane & 16) ? 8 : 0;
    const int u0      = wv * 16;

    for (int i = tid; i < 16 * HP; i += 256) hs[i] = (__bf16)0.f;

    // ---- encoder weight fragments (recurrent; + input if not precomputed) ----
    v16bf wf[FUSE_DENSE ? 24 : (USE_ZX ? 16 : 24)];
    if constexpr (!USE_ZX) {
#pragma unroll
        for (int g = 0; g < 4; ++g)
#pragma unroll
            for (int kt = 0; kt < 2; ++kt)
                wf[16 + g * 2 + kt] = load_w_frag(wpk, FRAG_ENC_K + (g * 8 + wv) * 2 + kt);
    }
#pragma unroll
    for (int g = 0; g < 4; ++g)
#pragma unroll
        for (int kt = 0; kt < 4; ++kt)
            wf[g * 4 + kt] = load_w_frag(wpk, FRAG_ENC_R + (g * 8 + wv) * 4 + kt);

    float bias[4];
#pragma unroll
    for (int g = 0; g < 4; ++g) bias[g] = enc_b[g * U_ + u0 + col];

    v8f c = splat8(0.f);
    v8bf zc[4];

    if constexpr (USE_ZX) {
#pragma unroll
        for (int g = 0; g < 4; ++g) {
            size_t fi = (((size_t)0 * 16 + bt) * 8 + wv) * 4 + g;
            zc[g] = *(const v8bf*)(zxp + fi * 256 + lane * 8);
        }
    } else {
        // prologue: stage x_0 into buffer 0
        const int i  = tid >> 4;
        const int k0 = (tid & 15) * 4;
        const float4 v = *(const float4*)(x + ((size_t)(b0 + i) * T_ + 0) * F_ + k0);
        __bf16* d = xs[0] + i * XP + k0;
        d[0] = (__bf16)v.x; d[1] = (__bf16)v.y;
        d[2] = (__bf16)v.z; d[3] = (__bf16)v.w;
    }

    // =============================== encoder ===============================
    for (int t = 0; t < T_; ++t) {
        v8bf zn[4];
        if constexpr (USE_ZX) {
            // prefetch next step's zx fragments (off the critical path)
            const int tn = (t + 1 < T_) ? t + 1 : t;
#pragma unroll
            for (int g = 0; g < 4; ++g) {
                size_t fi = (((size_t)tn * 16 + bt) * 8 + wv) * 4 + g;
                zn[g] = *(const v8bf*)(zxp + fi * 256 + lane * 8);
            }
        }
        __syncthreads();   // previous h writes + current x buffer visible
        if constexpr (!USE_ZX) {
            // stage x_{t+1} into the other buffer; overlaps this step's compute
            if (t + 1 < T_) {
                const int i  = tid >> 4;
                const int k0 = (tid & 15) * 4;
                const float4 v =
                    *(const float4*)(x + ((size_t)(b0 + i) * T_ + (t + 1)) * F_ + k0);
                __bf16* d = xs[(t + 1) & 1] + i * XP + k0;
                d[0] = (__bf16)v.x; d[1] = (__bf16)v.y;
                d[2] = (__bf16)v.z; d[3] = (__bf16)v.w;
            }
        }

        v8f acc[4];
        if constexpr (USE_ZX) {
#pragma unroll
            for (int g = 0; g < 4; ++g) {
                v8f a = splat8(bias[g]);
#pragma unroll
                for (int r = 0; r < 8; ++r) a[r] += (float)zc[g][r];
                acc[g] = a;
            }
        } else {
#pragma unroll
            for (int g = 0; g < 4; ++g) acc[g] = splat8(bias[g]);
#pragma unroll
            for (int kt = 0; kt < 2; ++kt) {          // x @ Wk (K = 64)
                v16bf a = load_a_frag(xs[t & 1], XP, kt);
#pragma unroll
                for (int g = 0; g < 4; ++g)
                    acc[g] = wmma_bf16(a, wf[16 + g * 2 + kt], acc[g]);
            }
        }
#pragma unroll
        for (int kt = 0; kt < 4; ++kt) {              // h @ Wr (K = 128)
            v16bf a = load_a_frag(hs, HP, kt);
#pragma unroll
            for (int g = 0; g < 4; ++g)
                acc[g] = wmma_bf16(a, wf[g * 4 + kt], acc[g]);
        }

        float hnew[8];
#pragma unroll
        for (int r = 0; r < 8; ++r) {
            float ig = sigmoidf_(acc[0][r]);
            float fg = sigmoidf_(acc[1][r]);
            float gg = tanhf_(acc[2][r]);
            float og = sigmoidf_(acc[3][r]);
            float cn = fg * c[r] + ig * gg;
            c[r]    = cn;
            hnew[r] = og * tanhf_(cn);
        }
        __syncthreads();   // all waves done reading hs
#pragma unroll
        for (int r = 0; r < 8; ++r)
            hs[(rowBase + r) * HP + u0 + col] = (__bf16)hnew[r];

        if constexpr (USE_ZX) {
#pragma unroll
            for (int g = 0; g < 4; ++g) zc[g] = zn[g];
        }
    }

    // ---- phase switch: decoder gate weights = dec_k + dec_rk (pre-summed) ----
#pragma unroll
    for (int g = 0; g < 4; ++g)
#pragma unroll
        for (int kt = 0; kt < 4; ++kt)
            wf[g * 4 + kt] = load_w_frag(wpk, FRAG_DEC_SUM + (g * 8 + wv) * 4 + kt);
    if constexpr (FUSE_DENSE) {
#pragma unroll
        for (int kt = 0; kt < 4; ++kt)
            wf[16 + kt] = load_w_frag(wpk, FRAG_W1 + wv * 4 + kt);
#pragma unroll
        for (int kt = 0; kt < 4; ++kt)
            wf[20 + kt] = load_w_frag(wpk, FRAG_W2 + (wv & 3) * 4 + kt);
    }
#pragma unroll
    for (int g = 0; g < 4; ++g) bias[g] = dec_b[g * U_ + u0 + col];
    const float bias1 = FUSE_DENSE ? b1[u0 + col] : 0.f;
    const float bias2 = FUSE_DENSE ? b2[(wv & 3) * 16 + col] : 0.f;

    // =============================== decoder ===============================
    for (int t = 0; t < T_; ++t) {
        __syncthreads();   // hs complete

        v8f acc[4];
#pragma unroll
        for (int g = 0; g < 4; ++g) acc[g] = splat8(bias[g]);
#pragma unroll
        for (int kt = 0; kt < 4; ++kt) {              // h @ (dec_k + dec_rk)
            v16bf a = load_a_frag(hs, HP, kt);
#pragma unroll
            for (int g = 0; g < 4; ++g)
                acc[g] = wmma_bf16(a, wf[g * 4 + kt], acc[g]);
        }

        float hnew[8];
#pragma unroll
        for (int r = 0; r < 8; ++r) {
            float ig = sigmoidf_(acc[0][r]);
            float fg = sigmoidf_(acc[1][r]);
            float gg = tanhf_(acc[2][r]);
            float og = sigmoidf_(acc[3][r]);
            float cn = fg * c[r] + ig * gg;
            c[r]    = cn;
            hnew[r] = og * tanhf_(cn);
        }
        __syncthreads();   // reads of hs done
#pragma unroll
        for (int r = 0; r < 8; ++r) {
            __bf16 hb = (__bf16)hnew[r];
            hs[(rowBase + r) * HP + u0 + col] = hb;
            if constexpr (!FUSE_DENSE) {
                // fire-and-forget scatter to seq[t][b][u]; not on the dep chain
                seq[(size_t)(t * B_ + b0 + rowBase + r) * U_ + u0 + col] = hb;
            }
        }

        if constexpr (FUSE_DENSE) {
            __syncthreads();   // new h tile complete
            {   // hidden = relu(h @ w1 + b1): wave w computes n-tile w
                v8f d1 = splat8(bias1);
#pragma unroll
                for (int kt = 0; kt < 4; ++kt) {
                    v16bf a = load_a_frag(hs, HP, kt);
                    d1 = wmma_bf16(a, wf[16 + kt], d1);
                }
#pragma unroll
                for (int r = 0; r < 8; ++r)
                    dsb[(rowBase + r) * HP + u0 + col] = (__bf16)fmaxf(d1[r], 0.f);
            }
            __syncthreads();   // hidden tile ready
            if (wv < 4) {      // out[:, t, :] = hidden @ w2 + b2
                v8f d2 = splat8(bias2);
#pragma unroll
                for (int kt = 0; kt < 4; ++kt) {
                    v16bf a = load_a_frag(dsb, HP, kt);
                    d2 = wmma_bf16(a, wf[20 + kt], d2);
                }
#pragma unroll
                for (int r = 0; r < 8; ++r)
                    out[((size_t)(b0 + rowBase + r) * T_ + t) * F_ + wv * 16 + col] = d2[r];
            }
        }
    }
}

// ---------------------------------------------------------------------------
// Kernel 4 (optional, full-chip): out = relu(seq @ w1 + b1) @ w2 + b2.
// seq rows r = t*B + b (bf16, row-major U=128). 32 rows per workgroup.
// ---------------------------------------------------------------------------
__global__ __launch_bounds__(256)
void dense_head_kernel(const __bf16* __restrict__ seq,
                       const __bf16* __restrict__ wpk,
                       const float* __restrict__ b1,
                       const float* __restrict__ b2,
                       float* __restrict__ out) {
    __shared__ __align__(16) __bf16 hid[32 * HP];
    const int tid     = threadIdx.x;
    const int lane    = tid & 31;
    const int wv      = tid >> 5;
    const int col     = lane & 15;
    const int rowBase = (lane & 16) ? 8 : 0;
    const int gr0     = blockIdx.x * 32;   // first seq row of this workgroup
    const int mt      = wv >> 2;           // 0..1 (row tile)

    // stage 1: hidden = relu(seq @ w1 + b1); wave handles 2 of 8 n-tiles
#pragma unroll
    for (int s = 0; s < 2; ++s) {
        const int nt = (wv & 3) * 2 + s;
        v8f acc = splat8(b1[nt * 16 + col]);
#pragma unroll
        for (int kt = 0; kt < 4; ++kt) {
            v16bf a = load_a_frag(seq + (size_t)(gr0 + mt * 16) * U_, U_, kt);
            acc = wmma_bf16(a, load_w_frag(wpk, FRAG_W1 + nt * 4 + kt), acc);
        }
#pragma unroll
        for (int r = 0; r < 8; ++r)
            hid[(mt * 16 + rowBase + r) * HP + nt * 16 + col] =
                (__bf16)fmaxf(acc[r], 0.f);
    }
    __syncthreads();

    // stage 2: out = hidden @ w2 + b2; one (mt, nt) tile per wave
    {
        const int nt = wv & 3;
        v8f acc = splat8(b2[nt * 16 + col]);
#pragma unroll
        for (int kt = 0; kt < 4; ++kt) {
            v16bf a = load_a_frag(hid + (size_t)mt * 16 * HP, HP, kt);
            acc = wmma_bf16(a, load_w_frag(wpk, FRAG_W2 + nt * 4 + kt), acc);
        }
#pragma unroll
        for (int r = 0; r < 8; ++r) {
            int gr = gr0 + mt * 16 + rowBase + r;
            int t  = gr >> 8;          // gr = t*256 + b
            int b  = gr & 255;
            out[((size_t)b * T_ + t) * F_ + nt * 16 + col] = acc[r];
        }
    }
}

// ---------------------------------------------------------------------------
extern "C" void kernel_launch(void* const* d_in, const int* in_sizes, int n_in,
                              void* d_out, int out_size, void* d_ws, size_t ws_size,
                              hipStream_t stream) {
    const float* x      = (const float*)d_in[0];
    const float* enc_k  = (const float*)d_in[1];
    const float* enc_rk = (const float*)d_in[2];
    const float* enc_b  = (const float*)d_in[3];
    const float* dec_k  = (const float*)d_in[4];
    const float* dec_rk = (const float*)d_in[5];
    const float* dec_b  = (const float*)d_in[6];
    const float* w1     = (const float*)d_in[7];
    const float* b1     = (const float*)d_in[8];
    const float* w2     = (const float*)d_in[9];
    const float* b2     = (const float*)d_in[10];
    float* out = (float*)d_out;

    // scratch layout (byte offsets)
    const size_t wpk_off  = 0;                                   // 368 KB
    const size_t seq_off  = (size_t)1 << 19;                     // 512 KB
    const size_t seq_len  = (size_t)T_ * B_ * U_ * 2;            // 32 MB
    const size_t zx_off   = seq_off + (((seq_len + 4095) / 4096) * 4096);
    const size_t zx_len   = (size_t)T_ * B_ * 4 * U_ * 2;        // 128 MB

    const bool have_seq = ws_size >= seq_off + seq_len;
    const bool have_zx  = have_seq && (ws_size >= zx_off + zx_len);

    __bf16* wpk = (__bf16*)((char*)d_ws + wpk_off);
    __bf16* seq = (__bf16*)((char*)d_ws + seq_off);
    __bf16* zxp = (__bf16*)((char*)d_ws + zx_off);

    pack_weights_kernel<<<(FRAG_TOTAL * 32 + 255) / 256, 256, 0, stream>>>(
        enc_k, enc_rk, dec_k, dec_rk, w1, w2, wpk);

    if (have_zx) {
        zx_precompute_kernel<<<dim3(T_, 16), 256, 0, stream>>>(x, wpk, zxp);
        lstm_encdec_kernel<true, false><<<B_ / 16, 256, 0, stream>>>(
            x, enc_b, dec_b, b1, b2, wpk, zxp, seq, out);
        dense_head_kernel<<<(T_ * B_) / 32, 256, 0, stream>>>(seq, wpk, b1, b2, out);
    } else if (have_seq) {
        lstm_encdec_kernel<false, false><<<B_ / 16, 256, 0, stream>>>(
            x, enc_b, dec_b, b1, b2, wpk, zxp, seq, out);
        dense_head_kernel<<<(T_ * B_) / 32, 256, 0, stream>>>(seq, wpk, b1, b2, out);
    } else {
        lstm_encdec_kernel<false, true><<<B_ / 16, 256, 0, stream>>>(
            x, enc_b, dec_b, b1, b2, wpk, zxp, seq, out);
    }
}
